// RegionProposalNetwork_86835648791321
// MI455X (gfx1250) — compile-verified, matrix operand order checked by reading, not run on hardware
//
#include <hip/hip_runtime.h>
#include <hip/hip_bf16.h>

typedef __attribute__((ext_vector_type(16))) _Float16 v16h;
typedef __attribute__((ext_vector_type(8)))  _Float16 v8h;
typedef __attribute__((ext_vector_type(8)))  float    v8f;

#define NIMG   8
#define CIN    512
#define COUT   512
#define WW     50
#define SP     2500          // 50*50
#define K1     4608          // 9*512  (k' = (ky*3+kx)*512 + ci)
#define NANCH  22500
#define NPRE   12000
#define NPOST  2000
#define NMS_TH 0.7f
#define MINSZ  16.0f

// d_out element offsets (float/int32 units)
#define OFF_LOCS   0
#define OFF_SCORES 720000
#define OFF_ROIS   1080000
#define OFF_RIDX   1160000
#define OFF_ANCH   1176000

__device__ __forceinline__ float clampf(float v, float lo, float hi) {
    return fminf(fmaxf(v, lo), hi);
}
__device__ __forceinline__ unsigned fkey(float f) {
    unsigned u = __float_as_uint(f);
    return (u & 0x80000000u) ? ~u : (u | 0x80000000u);
}
// CDNA5 async memory->LDS copy (per active lane, 16B), tracked by ASYNCcnt
__device__ __forceinline__ void async_ld_b128(unsigned lds_off, const void* gaddr) {
    asm volatile("global_load_async_to_lds_b128 %0, %1, off"
                 :: "v"(lds_off), "v"(gaddr) : "memory");
}
__device__ __forceinline__ void wait_async0() {
    asm volatile("s_wait_asynccnt 0" ::: "memory");
}

// ---------------- weight / input conversion ----------------
// w1h[co][k'] with k' = j*512 + ci  (j = ky*3+kx)
__global__ void k_convert_w1(const float* __restrict__ w, _Float16* __restrict__ wh) {
    int i = blockIdx.x * blockDim.x + threadIdx.x;
    if (i >= COUT * K1) return;
    int co = i / K1, rem = i - co * K1;
    int j = rem >> 9, ci = rem & 511;
    wh[i] = (_Float16)w[(size_t)co * K1 + ci * 9 + j];
}

// channel-last padded input: xpadT[n][yy][xx][ci], yy,xx in [0,52)
__global__ void k_pad_x(const float* __restrict__ x, _Float16* __restrict__ xp) {
    int i = blockIdx.x * blockDim.x + threadIdx.x;
    const int TOT = NIMG * 52 * 52 * CIN;
    if (i >= TOT) return;
    int ci = i & 511, r = i >> 9;
    int xx = r % 52, r2 = r / 52;
    int yy = r2 % 52, n = r2 / 52;
    float v = 0.f;
    if (yy >= 1 && yy <= 50 && xx >= 1 && xx <= 50)
        v = x[((size_t)(n * CIN + ci) * 50 + (yy - 1)) * 50 + (xx - 1)];
    xp[i] = (_Float16)v;
}

__global__ void k_convert_w2(const float* __restrict__ clsw, const float* __restrict__ regw,
                             _Float16* __restrict__ wh) {
    int i = blockIdx.x * blockDim.x + threadIdx.x;
    if (i >= 64 * 512) return;
    int co = i >> 9, ci = i & 511;
    float v = 0.f;
    if (co < 18)       v = clsw[co * 512 + ci];
    else if (co < 54)  v = regw[(co - 18) * 512 + ci];
    wh[i] = (_Float16)v;
}

// ---------------- anchors ----------------
__global__ void k_anchors(float* __restrict__ anch) {
    int i = blockIdx.x * blockDim.x + threadIdx.x;
    if (i >= NANCH) return;
    int pos = i / 9, a = i - pos * 9;
    int y = pos / WW, x = pos - y * WW;
    const float ratios[3] = {0.5f, 1.0f, 2.0f};
    const float scales[3] = {8.0f, 16.0f, 32.0f};
    float r = ratios[a / 3], s = scales[a % 3];
    float h = 32.f * s * sqrtf(r);
    float w = 32.f * s * sqrtf(1.0f / r);
    float sy = (float)(y * 32), sx = (float)(x * 32);
    anch[i * 4 + 0] = 16.f - 0.5f * h + sy;
    anch[i * 4 + 1] = 16.f - 0.5f * w + sx;
    anch[i * 4 + 2] = 16.f + 0.5f * h + sy;
    anch[i * 4 + 3] = 16.f + 0.5f * w + sx;
}

// ---------------- conv1: 3x3 SAME implicit GEMM, WMMA + async double-buffered LDS ----------
// block = 256 threads (8 waves, 4 co-subtiles x 2 sp-subtiles); block tile 64 Cout x 64 sp
// each wave: 16 Cout x 32 sp = 2 WMMAs sharing one A fragment
__global__ __launch_bounds__(256)
void k_conv1_wmma(const _Float16* __restrict__ w1h, const _Float16* __restrict__ xpadT,
                  const float* __restrict__ bias, _Float16* __restrict__ feat) {
    const int spBase = blockIdx.x * 64;
    const int coBase = blockIdx.y * 64;
    const int n      = blockIdx.z;
    const int tid  = threadIdx.x;
    const int lane = tid & 31, wave = tid >> 5;
    const int wr = wave & 3, wc = wave >> 2;

    __shared__ alignas(16) _Float16 As[2][64 * 40];   // [co][k], k-stride 40
    __shared__ alignas(16) _Float16 Bs[2][64 * 40];   // [sp][k], k-stride 40

    v8f acc0 = {}, acc1 = {};

    // A staging: 256 threads x 16B
    const int a_co = tid >> 2, a_k = (tid & 3) * 8;
    const _Float16* wrow = w1h + (size_t)(coBase + a_co) * K1 + a_k;
    // B staging: 256 threads x 16B (contiguous ci)
    const int b_sp = tid >> 2, b_k = (tid & 3) * 8;
    const int pos  = spBase + b_sp;
    const int py   = pos / WW, px = pos - py * WW;
    const _Float16* xin = xpadT + ((size_t)n * 52 * 52) * CIN
                        + (size_t)(py * 52 + px) * CIN + b_k;

    // fragment indices (ISA 16-bit A/B striping)
    const int halfk = (lane >> 4) * 8;
    const int arow  = wr * 16 + (lane & 15);
    const int bcol0 = wc * 32 + (lane & 15);
    const int bcol1 = bcol0 + 16;

    // zero-fill B rows for out-of-range spatial positions (persist across loop)
    if (pos >= SP) {
        v8h z = {};
        *(v8h*)&Bs[0][b_sp * 40 + b_k] = z;
        *(v8h*)&Bs[1][b_sp * 40 + b_k] = z;
    }

    auto issue = [&](int kb, int buf) {
        async_ld_b128((unsigned)(size_t)&As[buf][a_co * 40 + a_k], wrow + kb);
        if (pos < SP) {
            int j = kb >> 9, cib = kb & 511;
            int ky = (j * 11) >> 5;            // j/3 for j in [0,9)
            int kx = j - 3 * ky;
            int uoff = ((ky * 52 + kx) << 9) + cib;   // uniform per K-step
            async_ld_b128((unsigned)(size_t)&Bs[buf][b_sp * 40 + b_k], xin + uoff);
        }
    };

    issue(0, 0);
    int buf = 0;
    for (int kb = 0; kb < K1; kb += 32) {
        wait_async0();
        __syncthreads();
        if (kb + 32 < K1) issue(kb + 32, buf ^ 1);

        v8h alo = *(const v8h*)&As[buf][arow * 40 + halfk];
        v8h ahi = *(const v8h*)&As[buf][arow * 40 + 16 + halfk];
        v8h b0l = *(const v8h*)&Bs[buf][bcol0 * 40 + halfk];
        v8h b0h = *(const v8h*)&Bs[buf][bcol0 * 40 + 16 + halfk];
        v8h b1l = *(const v8h*)&Bs[buf][bcol1 * 40 + halfk];
        v8h b1h = *(const v8h*)&Bs[buf][bcol1 * 40 + 16 + halfk];
        v16h af, bf0, bf1;
        #pragma unroll
        for (int e = 0; e < 8; ++e) {
            af[e]  = alo[e]; af[e + 8]  = ahi[e];
            bf0[e] = b0l[e]; bf0[e + 8] = b0h[e];
            bf1[e] = b1l[e]; bf1[e + 8] = b1h[e];
        }
        acc0 = __builtin_amdgcn_wmma_f32_16x16x32_f16(false, af, false, bf0,
                                                      (short)0, acc0, false, false);
        acc1 = __builtin_amdgcn_wmma_f32_16x16x32_f16(false, af, false, bf1,
                                                      (short)0, acc1, false, false);
        buf ^= 1;
    }

    const int sp0 = spBase + wc * 32 + (lane & 15);
    const int sp1 = sp0 + 16;
    const int co8 = coBase + wr * 16 + (lane >> 4) * 8;
    float bb[8];
    #pragma unroll
    for (int v = 0; v < 8; ++v) bb[v] = bias[co8 + v];
    if (sp0 < SP) {
        v8h fr;
        #pragma unroll
        for (int v = 0; v < 8; ++v) fr[v] = (_Float16)fmaxf(acc0[v] + bb[v], 0.f);
        *(v8h*)&feat[((size_t)n * SP + sp0) * CIN + co8] = fr;
    }
    if (sp1 < SP) {
        v8h fr;
        #pragma unroll
        for (int v = 0; v < 8; ++v) fr[v] = (_Float16)fmaxf(acc1[v] + bb[v], 0.f);
        *(v8h*)&feat[((size_t)n * SP + sp1) * CIN + co8] = fr;
    }
}

// ---------------- conv2: fused 1x1 cls(18)+reg(36) padded to 64 rows ----------------
__global__ __launch_bounds__(256)
void k_conv2_wmma(const _Float16* __restrict__ w2h, const _Float16* __restrict__ feat,
                  const float* __restrict__ clsb, const float* __restrict__ regb,
                  float* __restrict__ out2) {
    const int spBase = blockIdx.x * 64;
    const int n      = blockIdx.y;
    const int tid  = threadIdx.x;
    const int lane = tid & 31, wave = tid >> 5;
    const int wr = wave & 3, wc = wave >> 2;

    __shared__ alignas(16) _Float16 As[2][64 * 40];
    __shared__ alignas(16) _Float16 Bs[2][64 * 40];

    v8f acc0 = {}, acc1 = {};
    const int a_co = tid >> 2, a_k = (tid & 3) * 8;
    const int b_sp = tid >> 2, b_k = (tid & 3) * 8;
    const int pos  = spBase + b_sp;
    const _Float16* frow = feat + ((size_t)n * SP + pos) * CIN + b_k;
    const int halfk = (lane >> 4) * 8;
    const int arow  = wr * 16 + (lane & 15);
    const int bcol0 = wc * 32 + (lane & 15);
    const int bcol1 = bcol0 + 16;

    if (pos >= SP) {
        v8h z = {};
        *(v8h*)&Bs[0][b_sp * 40 + b_k] = z;
        *(v8h*)&Bs[1][b_sp * 40 + b_k] = z;
    }

    auto issue = [&](int kb, int buf) {
        async_ld_b128((unsigned)(size_t)&As[buf][a_co * 40 + a_k],
                      w2h + (size_t)a_co * CIN + kb + a_k);
        if (pos < SP)
            async_ld_b128((unsigned)(size_t)&Bs[buf][b_sp * 40 + b_k], frow + kb);
    };

    issue(0, 0);
    int buf = 0;
    for (int kb = 0; kb < CIN; kb += 32) {
        wait_async0();
        __syncthreads();
        if (kb + 32 < CIN) issue(kb + 32, buf ^ 1);

        v8h alo = *(const v8h*)&As[buf][arow * 40 + halfk];
        v8h ahi = *(const v8h*)&As[buf][arow * 40 + 16 + halfk];
        v8h b0l = *(const v8h*)&Bs[buf][bcol0 * 40 + halfk];
        v8h b0h = *(const v8h*)&Bs[buf][bcol0 * 40 + 16 + halfk];
        v8h b1l = *(const v8h*)&Bs[buf][bcol1 * 40 + halfk];
        v8h b1h = *(const v8h*)&Bs[buf][bcol1 * 40 + 16 + halfk];
        v16h af, bf0, bf1;
        #pragma unroll
        for (int e = 0; e < 8; ++e) {
            af[e]  = alo[e]; af[e + 8]  = ahi[e];
            bf0[e] = b0l[e]; bf0[e + 8] = b0h[e];
            bf1[e] = b1l[e]; bf1[e + 8] = b1h[e];
        }
        acc0 = __builtin_amdgcn_wmma_f32_16x16x32_f16(false, af, false, bf0,
                                                      (short)0, acc0, false, false);
        acc1 = __builtin_amdgcn_wmma_f32_16x16x32_f16(false, af, false, bf1,
                                                      (short)0, acc1, false, false);
        buf ^= 1;
    }

    const int sp0 = spBase + wc * 32 + (lane & 15);
    const int sp1 = sp0 + 16;
    #pragma unroll
    for (int v = 0; v < 8; ++v) {
        int co2 = wr * 16 + (lane >> 4) * 8 + v;
        if (co2 < 54) {
            float b = (co2 < 18) ? clsb[co2] : regb[co2 - 18];
            if (sp0 < SP) out2[((size_t)n * 54 + co2) * SP + sp0] = acc0[v] + b;
            if (sp1 < SP) out2[((size_t)n * 54 + co2) * SP + sp1] = acc1[v] + b;
        }
    }
}

// ---------------- postprocess ----------------
__global__ void k_post(const float* __restrict__ out2, const float* __restrict__ anch,
                       float* __restrict__ rlocs, float* __restrict__ rscores,
                       float* __restrict__ roi, float* __restrict__ scf,
                       const int* __restrict__ img_h, const int* __restrict__ img_w) {
    int idx = blockIdx.x * blockDim.x + threadIdx.x;
    if (idx >= NIMG * NANCH) return;
    int n = idx / NANCH, i = idx - n * NANCH;
    const float* o2 = out2 + (size_t)n * 54 * SP;
    const float* locsf = o2 + 18 * SP;
    float lv[4];
    #pragma unroll
    for (int j = 0; j < 4; ++j) {
        int m = 4 * i + j;
        float v = locsf[(m / SP) * SP + (m % SP)];
        rlocs[(size_t)(n * NANCH + i) * 4 + j] = v;
        lv[j] = v;
    }
    float sv[2];
    #pragma unroll
    for (int j = 0; j < 2; ++j) {
        int m = 2 * i + j;
        float s = o2[(m / SP) * SP + (m % SP)];
        rscores[(size_t)(n * NANCH + i) * 2 + j] = s;
        sv[j] = s;
    }
    float mx = fmaxf(sv[0], sv[1]);
    float e0 = expf(sv[0] - mx), e1 = expf(sv[1] - mx);
    float fg = e1 / (e0 + e1);

    float a0 = anch[i * 4 + 0], a1 = anch[i * 4 + 1];
    float a2 = anch[i * 4 + 2], a3 = anch[i * 4 + 3];
    float h = a2 - a0, w = a3 - a1;
    float cy = a0 + 0.5f * h, cx = a1 + 0.5f * w;
    float cty = lv[0] * h + cy, ctx = lv[1] * w + cx;
    float hh = expf(lv[2]) * h, wwv = expf(lv[3]) * w;
    float IH = (float)img_h[0], IW = (float)img_w[0];
    float r0 = clampf(cty - 0.5f * hh, 0.f, IH);
    float r1 = clampf(ctx - 0.5f * wwv, 0.f, IW);
    float r2 = clampf(cty + 0.5f * hh, 0.f, IH);
    float r3 = clampf(ctx + 0.5f * wwv, 0.f, IW);
    float* rr = roi + (size_t)(n * NANCH + i) * 4;
    rr[0] = r0; rr[1] = r1; rr[2] = r2; rr[3] = r3;
    float hs = r2 - r0, wsz = r3 - r1;
    scf[(size_t)n * NANCH + i] = (hs >= MINSZ && wsz >= MINSZ) ? fg : -__builtin_inff();
}

// ---------------- deterministic top-12000 per image ----------------
__global__ __launch_bounds__(1024)
void k_topk(const float* __restrict__ scf, const float* __restrict__ roi,
            float* __restrict__ selb, float* __restrict__ sels, float* __restrict__ area) {
    const int n = blockIdx.x, tid = threadIdx.x;
    const float* s = scf + (size_t)n * NANCH;
    const float* r = roi + (size_t)n * NANCH * 4;
    float* ob = selb + (size_t)n * NPRE * 4;
    float* os = sels + (size_t)n * NPRE;
    float* oa = area + (size_t)n * NPRE;

    __shared__ int sred[1024];
    __shared__ unsigned sCur;
    __shared__ int sCntG, sGRun, sERun;
    if (tid == 0) sCur = 0u;
    __syncthreads();

    for (int bit = 31; bit >= 0; --bit) {
        unsigned cand = sCur | (1u << bit);
        int cnt = 0;
        for (int i = tid; i < NANCH; i += 1024) cnt += (fkey(s[i]) >= cand) ? 1 : 0;
        sred[tid] = cnt; __syncthreads();
        for (int st = 512; st > 0; st >>= 1) {
            if (tid < st) sred[tid] += sred[tid + st];
            __syncthreads();
        }
        if (tid == 0 && sred[0] >= NPRE) sCur = cand;
        __syncthreads();
    }
    const unsigned thr = sCur;
    {
        int cnt = 0;
        for (int i = tid; i < NANCH; i += 1024) cnt += (fkey(s[i]) > thr) ? 1 : 0;
        sred[tid] = cnt; __syncthreads();
        for (int st = 512; st > 0; st >>= 1) {
            if (tid < st) sred[tid] += sred[tid + st];
            __syncthreads();
        }
        if (tid == 0) { sCntG = sred[0]; sGRun = 0; sERun = 0; }
        __syncthreads();
    }
    const int cntG = sCntG;

    for (int base = 0; base < NANCH; base += 1024) {
        int i = base + tid;
        bool valid = i < NANCH;
        unsigned k = valid ? fkey(s[i]) : 0u;
        int fG = (valid && k > thr) ? 1 : 0;
        int fE = (valid && k == thr) ? 1 : 0;
        int gBase = sGRun, eBase = sERun;

        sred[tid] = fG; __syncthreads();
        for (int off = 1; off < 1024; off <<= 1) {
            int v = (tid >= off) ? sred[tid - off] : 0;
            __syncthreads();
            sred[tid] += v;
            __syncthreads();
        }
        int gExcl = sred[tid] - fG, gTot = sred[1023];
        __syncthreads();

        sred[tid] = fE; __syncthreads();
        for (int off = 1; off < 1024; off <<= 1) {
            int v = (tid >= off) ? sred[tid - off] : 0;
            __syncthreads();
            sred[tid] += v;
            __syncthreads();
        }
        int eExcl = sred[tid] - fE, eTot = sred[1023];
        __syncthreads();

        int p = -1;
        if (fG) p = gBase + gExcl;
        else if (fE) p = cntG + eBase + eExcl;
        if (p >= 0 && p < NPRE) {
            float b0 = r[(size_t)i * 4 + 0], b1 = r[(size_t)i * 4 + 1];
            float b2 = r[(size_t)i * 4 + 2], b3 = r[(size_t)i * 4 + 3];
            ob[(size_t)p * 4 + 0] = b0; ob[(size_t)p * 4 + 1] = b1;
            ob[(size_t)p * 4 + 2] = b2; ob[(size_t)p * 4 + 3] = b3;
            os[p] = s[i];
            oa[p] = (b2 - b0) * (b3 - b1);
        }
        if (tid == 0) { sGRun = gBase + gTot; sERun = eBase + eTot; }
        __syncthreads();
    }
}

// ---------------- NMS ----------------
__global__ __launch_bounds__(1024)
void k_nms(const float* __restrict__ selb, float* __restrict__ sels,
           const float* __restrict__ area, int* __restrict__ keep,
           float* __restrict__ rois_out, int* __restrict__ ridx_out) {
    const int n = blockIdx.x, tid = threadIdx.x;
    const float* ob = selb + (size_t)n * NPRE * 4;
    float* os = sels + (size_t)n * NPRE;
    const float* oa = area + (size_t)n * NPRE;
    int* kp = keep + (size_t)n * NPOST;

    __shared__ float rv[1024];
    __shared__ int   ri[1024];
    __shared__ float bj[5];

    for (int it = 0; it < NPOST; ++it) {
        float bv = -__builtin_inff();
        int bi = 0x7fffffff;
        for (int i = tid; i < NPRE; i += 1024) {
            float v = os[i];
            if (v > bv || (v == bv && i < bi)) { bv = v; bi = i; }
        }
        rv[tid] = bv; ri[tid] = bi;
        __syncthreads();
        for (int st = 512; st > 0; st >>= 1) {
            if (tid < st) {
                if (rv[tid + st] > rv[tid] ||
                    (rv[tid + st] == rv[tid] && ri[tid + st] < ri[tid])) {
                    rv[tid] = rv[tid + st]; ri[tid] = ri[tid + st];
                }
            }
            __syncthreads();
        }
        if (tid == 0) {
            int j = ri[0];
            kp[it] = j;
            bj[0] = ob[(size_t)j * 4 + 0]; bj[1] = ob[(size_t)j * 4 + 1];
            bj[2] = ob[(size_t)j * 4 + 2]; bj[3] = ob[(size_t)j * 4 + 3];
            bj[4] = oa[j];
        }
        __syncthreads();
        float j0 = bj[0], j1 = bj[1], j2 = bj[2], j3 = bj[3], ja = bj[4];
        for (int i = tid; i < NPRE; i += 1024) {
            float yy1 = fmaxf(j0, ob[(size_t)i * 4 + 0]);
            float xx1 = fmaxf(j1, ob[(size_t)i * 4 + 1]);
            float yy2 = fminf(j2, ob[(size_t)i * 4 + 2]);
            float xx2 = fminf(j3, ob[(size_t)i * 4 + 3]);
            float inter = fmaxf(yy2 - yy1, 0.f) * fmaxf(xx2 - xx1, 0.f);
            float iou = inter / (oa[i] + ja - inter + 1e-9f);
            if (iou > NMS_TH) os[i] = -__builtin_inff();
        }
        __syncthreads();
    }
    for (int it = tid; it < NPOST; it += 1024) {
        int j = kp[it];
        float* row = rois_out + (size_t)(n * NPOST + it) * 5;
        row[0] = (float)n;
        row[1] = ob[(size_t)j * 4 + 0];
        row[2] = ob[(size_t)j * 4 + 1];
        row[3] = ob[(size_t)j * 4 + 2];
        row[4] = ob[(size_t)j * 4 + 3];
        ridx_out[n * NPOST + it] = n;
    }
}

extern "C" void kernel_launch(void* const* d_in, const int* in_sizes, int n_in,
                              void* d_out, int out_size, void* d_ws, size_t ws_size,
                              hipStream_t stream) {
    const float* x       = (const float*)d_in[0];
    const float* conv1_w = (const float*)d_in[1];
    const float* conv1_b = (const float*)d_in[2];
    const float* cls_w   = (const float*)d_in[3];
    const float* cls_b   = (const float*)d_in[4];
    const float* reg_w   = (const float*)d_in[5];
    const float* reg_b   = (const float*)d_in[6];
    const int*   img_h   = (const int*)d_in[7];
    const int*   img_w   = (const int*)d_in[8];

    char* ws = (char*)d_ws;
    size_t off = 0;
    auto take = [&](size_t bytes) -> char* {
        char* p = ws + off;
        off += bytes;
        off = (off + 255) & ~(size_t)255;
        return p;
    };
    _Float16* w1h  = (_Float16*)take((size_t)COUT * K1 * 2);
    _Float16* xpad = (_Float16*)take((size_t)NIMG * 52 * 52 * CIN * 2);
    _Float16* feat = (_Float16*)take((size_t)NIMG * SP * CIN * 2);
    _Float16* w2h  = (_Float16*)take((size_t)64 * 512 * 2);
    float*    out2 = (float*)take((size_t)NIMG * 54 * SP * 4);
    float*    roi  = (float*)take((size_t)NIMG * NANCH * 4 * 4);
    float*    scf  = (float*)take((size_t)NIMG * NANCH * 4);
    float*    selb = (float*)take((size_t)NIMG * NPRE * 4 * 4);
    float*    sels = (float*)take((size_t)NIMG * NPRE * 4);
    float*    area = (float*)take((size_t)NIMG * NPRE * 4);
    int*      keep = (int*)take((size_t)NIMG * NPOST * 4);

    float* o       = (float*)d_out;
    float* rlocs   = o + OFF_LOCS;
    float* rscores = o + OFF_SCORES;
    float* rois    = o + OFF_ROIS;
    int*   ridx    = (int*)d_out + OFF_RIDX;
    float* anch    = o + OFF_ANCH;

    k_convert_w1<<<(COUT * K1 + 255) / 256, 256, 0, stream>>>(conv1_w, w1h);
    k_pad_x<<<(NIMG * 52 * 52 * CIN + 255) / 256, 256, 0, stream>>>(x, xpad);
    k_convert_w2<<<(64 * 512 + 255) / 256, 256, 0, stream>>>(cls_w, reg_w, w2h);
    k_anchors<<<(NANCH + 255) / 256, 256, 0, stream>>>(anch);

    dim3 g1(40, COUT / 64, NIMG);                 // 64-sp tiles
    k_conv1_wmma<<<g1, 256, 0, stream>>>(w1h, xpad, conv1_b, feat);

    dim3 g2(40, NIMG);
    k_conv2_wmma<<<g2, 256, 0, stream>>>(w2h, feat, cls_b, reg_b, out2);

    k_post<<<(NIMG * NANCH + 255) / 256, 256, 0, stream>>>(out2, anch, rlocs, rscores,
                                                           roi, scf, img_h, img_w);
    k_topk<<<NIMG, 1024, 0, stream>>>(scf, roi, selb, sels, area);
    k_nms<<<NIMG, 1024, 0, stream>>>(selb, sels, area, keep, rois, ridx);
}
